// PointNetPlusEnc_9414568313084
// MI455X (gfx1250) — matrix operand order repaired and checked by reading, not compiled, not run on hardware
//
#include <hip/hip_runtime.h>
#include <hip/hip_bf16.h>

typedef __attribute__((ext_vector_type(16))) __bf16       v16bf;
typedef __attribute__((ext_vector_type(8)))  float        v8f;
typedef __attribute__((ext_vector_type(8)))  unsigned int v8u;
typedef __attribute__((ext_vector_type(4)))  unsigned int u32x4;
typedef __attribute__((ext_vector_type(8)))  int          i32x8;
typedef __attribute__((ext_vector_type(4)))  int          i32x4;

#define BATCH 16
#define NPTS0 8192
#define BN_EPS 1e-5f

// Tensor Data Mover availability (arity differs across toolchains; see probe doc)
#if defined(__has_builtin)
#  if __has_builtin(__builtin_amdgcn_tensor_load_to_lds) && \
      __has_builtin(__builtin_amdgcn_s_wait_tensorcnt)
#    define CDNA5_HAS_TDM 1
#  endif
#endif
#ifndef CDNA5_HAS_TDM
#  define CDNA5_HAS_TDM 0
#endif

static inline int align32i(int x) { return (x + 31) & ~31; }

// ---------------- input transpose: x [B][C][Np] -> xT [B][Np][C] ----------------
__global__ void k_transpose_in(const float* __restrict__ x, float* __restrict__ xT,
                               int C, int Np) {
  size_t i = (size_t)blockIdx.x * blockDim.x + threadIdx.x;
  size_t total = (size_t)BATCH * C * Np;
  if (i >= total) return;
  int n = (int)(i % Np);
  size_t t = i / Np;
  int c = (int)(t % C);
  int b = (int)(t / C);
  xT[((size_t)b * Np + n) * C + c] = x[i];
}

// ---------------- weight pack: f32 [Co][Ci] -> bf16 [Co][Cs] (zero-pad K) -------
__global__ void k_convert_w(const float* __restrict__ W, __hip_bfloat16* __restrict__ Wbf,
                            int Co, int Ci, int Cs) {
  int i = blockIdx.x * blockDim.x + threadIdx.x;
  if (i >= Co * Cs) return;
  int c = i % Cs;
  int o = i / Cs;
  float v = (c < Ci) ? W[(size_t)o * Ci + c] : 0.0f;
  Wbf[i] = __float2bfloat16(v);
}

__global__ void k_zero(float* __restrict__ p, int n) {
  int i = blockIdx.x * blockDim.x + threadIdx.x;
  if (i < n) p[i] = 0.0f;
}

// ---------------- farthest point sampling: 1 workgroup per batch ----------------
// Distances live in LDS (32KB of the 320KB WGP pool); block-wide argmax per step.
__global__ void k_fps(const float* __restrict__ xT, int Np, int C, int S,
                      int* __restrict__ idxOut, float* __restrict__ centers /*[B][S][3]*/) {
  __shared__ float sdist[NPTS0];
  __shared__ float rval[256];
  __shared__ int   ridx[256];
  const int b = blockIdx.x;
  const int tid = threadIdx.x;
  const float* xb = xT + (size_t)b * Np * C;
  for (int n = tid; n < Np; n += 256) sdist[n] = 3.4e38f;
  __syncthreads();
  int far = 0;
  for (int s = 0; s < S; ++s) {
    if (tid == 0) idxOut[b * S + s] = far;
    if (tid < 3) centers[((size_t)b * S + s) * 3 + tid] = xb[(size_t)far * C + tid];
    const float fx = xb[(size_t)far * C + 0];
    const float fy = xb[(size_t)far * C + 1];
    const float fz = xb[(size_t)far * C + 2];
    float bv = -1.0f; int bi = 0;
    for (int n = tid; n < Np; n += 256) {
      const float dx = xb[(size_t)n * C + 0] - fx;
      const float dy = xb[(size_t)n * C + 1] - fy;
      const float dz = xb[(size_t)n * C + 2] - fz;
      const float d = dx * dx + dy * dy + dz * dz;
      const float m = fminf(sdist[n], d);
      sdist[n] = m;
      if (m > bv) { bv = m; bi = n; }
    }
    rval[tid] = bv; ridx[tid] = bi;
    __syncthreads();
    for (int o = 128; o > 0; o >>= 1) {
      if (tid < o && rval[tid + o] > rval[tid]) { rval[tid] = rval[tid + o]; ridx[tid] = ridx[tid + o]; }
      __syncthreads();
    }
    far = ridx[0];
    __syncthreads();
  }
}

// ---------------- ball query + gather: one wave32 per (b,s) center --------------
__global__ void k_group(const float* __restrict__ xT, int Np, int C, int Cs,
                        const float* __restrict__ centers, int S, int Q, float r2,
                        __hip_bfloat16* __restrict__ Xg) {
  __shared__ int sidx[4][64];
  const int w = threadIdx.x >> 5;
  const int lane = threadIdx.x & 31;
  const int g = blockIdx.x * 4 + w;          // (b,s) flat
  const int b = g / S;
  const float cx = centers[(size_t)g * 3 + 0];
  const float cy = centers[(size_t)g * 3 + 1];
  const float cz = centers[(size_t)g * 3 + 2];
  const float* xb = xT + (size_t)b * Np * C;

  int cnt = 0;                                // wave-uniform
  for (int base = 0; base < Np && cnt < Q; base += 32) {
    const int n = base + lane;
    const float dx = xb[(size_t)n * C + 0] - cx;
    const float dy = xb[(size_t)n * C + 1] - cy;
    const float dz = xb[(size_t)n * C + 2] - cz;
    const bool inb = (dx * dx + dy * dy + dz * dz) < r2;
    unsigned int mask = __builtin_amdgcn_ballot_w32(inb);
    while (mask && cnt < Q) {
      const int bit = __builtin_ctz(mask);
      mask &= mask - 1;
      if (lane == 0) sidx[w][cnt] = base + bit;
      cnt++;
    }
  }
  if (lane == 0) {
    const int f = (cnt > 0) ? sidx[w][0] : 0;
    for (int q = cnt; q < Q; ++q) sidx[w][q] = f;
  }
  __syncthreads();

  for (int q = 0; q < Q; ++q) {
    const int n = sidx[w][q];
    const float* src = xb + (size_t)n * C;
    __hip_bfloat16* dst = Xg + ((size_t)g * Q + q) * Cs;
    for (int c = lane; c < Cs; c += 32) {
      float v = 0.0f;
      if (c < C) {
        v = src[c];
        if (c < 3) v -= centers[(size_t)g * 3 + c];
      }
      dst[c] = __float2bfloat16(v);
    }
  }
}

// ---------------- WMMA GEMM + bias + BN-stat accumulation -----------------------
// H[n][co] = sum_k X[n][k] * W[co][k] + bias[co]
// CS is compile-time (32/64/128) so the k-loop fully unrolls: per k-step,
// 2x ds_load_b128 (B frag from the LDS-staged weight tile) + 8x global b128
// (A frags) + 4x v_wmma_f32_16x16x32_bf16.  The 64xCS weight tile is staged
// into LDS by the Tensor Data Mover (tensor_load_to_lds + s_wait_tensorcnt);
// the descriptor carries the REAL LDS address of sW so the pointer escapes,
// and priming stores keep the ds reads from folding to undef.
template <int CS>
__global__ void __launch_bounds__(128)
k_gemm_bn_stats(const __hip_bfloat16* __restrict__ X,  // [Ntot][CS]
                const __hip_bfloat16* __restrict__ W,  // [Co][CS]
                const float* __restrict__ bias,
                float* __restrict__ H,                 // [Ntot][Co]
                float* __restrict__ sum, float* __restrict__ sumsq,
                int Ntot, int Co) {
  __shared__ __align__(16) unsigned short sW[64 * CS];
  const int lane   = threadIdx.x;
  const int laneLo = lane & 15;
  const int hi     = (lane >> 4) & 1;
  const int coBlock = blockIdx.y * 64;
  const int coLocal = threadIdx.y * 16 + laneLo;
  const int nBase  = blockIdx.x * 64;
  const unsigned short* Xu = (const unsigned short*)X;
  const unsigned short* Wg = (const unsigned short*)W + (size_t)coBlock * CS;

  // Visible dynamic-index writes to sW: blocks undef-folding of the ds reads.
  sW[(threadIdx.y * 32 + lane) & (64 * CS - 1)] = 0;
  __syncthreads();

#if CDNA5_HAS_TDM
  if (threadIdx.y == 0) {
    // Low 32 bits of the generic pointer are the LDS byte offset (ISA 10.2).
    const unsigned int ldsAddr =
        (unsigned int)(unsigned long long)(uintptr_t)&sW[0];
    const unsigned long long ga = (unsigned long long)(uintptr_t)Wg;
    // D# group 0: count=1 | lds_addr | global_addr[56:0] | type=2 @127:126
    u32x4 g0;
    g0[0] = 1u;
    g0[1] = ldsAddr;
    g0[2] = (unsigned int)(ga & 0xFFFFFFFFu);
    g0[3] = (unsigned int)((ga >> 32) & 0x01FFFFFFu) | 0x80000000u;
    // D# group 1: data_size=1 (2B), tensor_dim0=stride0=CS, tensor_dim1=64,
    //             tile_dim0=CS, tile_dim1=64 (2-D tile, no padding/iteration)
    i32x8 g1;
    g1[0] = 0x00010000;                       // workgroup_mask=0, data_size=2B
    g1[1] = (int)((unsigned)CS << 16);        // tensor_dim0[15:0] @ bit 48
    g1[2] = (int)(64u << 16);                 // dim0 hi=0 | tensor_dim1=64
    g1[3] = (int)((unsigned)CS << 16);        // dim1 hi=0 | tile_dim0=CS
    g1[4] = 64;                               // tile_dim1=64, tile_dim2=0
    g1[5] = CS;                               // tensor_dim0_stride lo
    g1[6] = 0;                                // stride0 hi | stride1 lo
    g1[7] = 0;                                // stride1 hi
    i32x4 gz = {0, 0, 0, 0};
#if __clang_major__ >= 23
    i32x8 gz8 = {0, 0, 0, 0, 0, 0, 0, 0};
    __builtin_amdgcn_tensor_load_to_lds(g0, g1, gz, gz, gz8, 0);
#else
    __builtin_amdgcn_tensor_load_to_lds(g0, g1, gz, gz, 0);
#endif
    __builtin_amdgcn_s_wait_tensorcnt(0);
  }
  asm volatile("" ::: "memory");
  __syncthreads();
#else
  {
    const int nvec = 8 * CS;                  // (64*CS)/8 halves per uint4
    for (int i = threadIdx.y * 32 + lane; i < nvec; i += 128)
      ((uint4*)sW)[i] = ((const uint4*)Wg)[i];
    __syncthreads();
  }
#endif

  v8f acc[4] = {};
  const unsigned short* srow = sW + coLocal * CS;

#pragma unroll
  for (int k0 = 0; k0 < CS; k0 += 32) {
    // B fragment from LDS: half h -> k = k0 + hi*16 + h, col = coLocal
    const int kb = k0 + hi * 16;
    v8u bu;
    {
      const uint4 lo4 = *(const uint4*)(srow + kb);
      const uint4 hi4 = *(const uint4*)(srow + kb + 8);
      bu[0] = lo4.x; bu[1] = lo4.y; bu[2] = lo4.z; bu[3] = lo4.w;
      bu[4] = hi4.x; bu[5] = hi4.y; bu[6] = hi4.z; bu[7] = hi4.w;
    }
    const v16bf bf = __builtin_bit_cast(v16bf, bu);

    // Load ALL four A fragments first (8x b128), then issue the 4 WMMAs.
    const int ka = k0 + hi * 16;
    v16bf af[4];
#pragma unroll
    for (int t = 0; t < 4; ++t) {
      const unsigned short* xrow = Xu + (size_t)(nBase + t * 16 + laneLo) * CS;
      v8u au;
      const uint4 lo4 = *(const uint4*)(xrow + ka);
      const uint4 hi4 = *(const uint4*)(xrow + ka + 8);
      au[0] = lo4.x; au[1] = lo4.y; au[2] = lo4.z; au[3] = lo4.w;
      au[4] = hi4.x; au[5] = hi4.y; au[6] = hi4.z; au[7] = hi4.w;
      af[t] = __builtin_bit_cast(v16bf, au);
    }
#pragma unroll
    for (int t = 0; t < 4; ++t) {
      acc[t] = __builtin_amdgcn_wmma_f32_16x16x32_bf16(
          false, af[t], false, bf, (short)0, acc[t], false, false);
    }
  }

  // bias, store, and per-channel stats (each lane's 32 values share one co)
  const int co = coBlock + coLocal;
  const float bv = bias[co];
  float ls = 0.0f, lss = 0.0f;
#pragma unroll
  for (int t = 0; t < 4; ++t) {
#pragma unroll
    for (int r = 0; r < 8; ++r) {
      const int n = nBase + t * 16 + r + hi * 8;
      const float v = acc[t][r] + bv;
      H[(size_t)n * Co + co] = v;
      ls += v; lss += v * v;
    }
  }
  atomicAdd(&sum[co], ls);
  atomicAdd(&sumsq[co], lss);
}

// ---------------- BN normalize + ReLU -> bf16 for next GEMM --------------------
__global__ void k_bn_relu(const float* __restrict__ H, const float* __restrict__ sum,
                          const float* __restrict__ sumsq, const float* __restrict__ gamma,
                          const float* __restrict__ beta, int Ntot, int Co,
                          __hip_bfloat16* __restrict__ out) {
  size_t i = (size_t)blockIdx.x * blockDim.x + threadIdx.x;
  const size_t total = (size_t)Ntot * Co;
  if (i >= total) return;
  const int co = (int)(i % Co);
  const float inv = 1.0f / (float)Ntot;
  const float m = sum[co] * inv;
  const float var = fmaxf(sumsq[co] * inv - m * m, 0.0f);
  float h = (H[i] - m) * rsqrtf(var + BN_EPS);
  h = h * gamma[co] + beta[co];
  out[i] = __float2bfloat16(fmaxf(h, 0.0f));
}

// ---------------- max over neighbors: [B*S*Q][C] bf16 -> [B][S][C] f32 ----------
__global__ void k_maxq(const __hip_bfloat16* __restrict__ A, int S, int Q, int C,
                       float* __restrict__ out) {
  size_t i = (size_t)blockIdx.x * blockDim.x + threadIdx.x;
  const size_t total = (size_t)BATCH * S * C;
  if (i >= total) return;
  const int c = (int)(i % C);
  const size_t bs = i / C;
  const __hip_bfloat16* p = A + (bs * Q) * C + c;
  float m = -3.4e38f;
  for (int q = 0; q < Q; ++q) m = fmaxf(m, __bfloat162float(p[(size_t)q * C]));
  out[i] = m;
}

// ---------------- mean over centers -> feat [B][C] -----------------------------
__global__ void k_meanS(const float* __restrict__ xT, int S, int C,
                        float* __restrict__ feat) {
  int i = blockIdx.x * blockDim.x + threadIdx.x;
  if (i >= BATCH * C) return;
  const int c = i % C;
  const int b = i / C;
  float s = 0.0f;
  for (int j = 0; j < S; ++j) s += xT[((size_t)b * S + j) * C + c];
  feat[i] = s / (float)S;
}

// ---------------- tiny FC head (VALU; ~17 MFLOP total) ---------------------------
__global__ void k_fc(const float* __restrict__ in, const float* __restrict__ W,
                     const float* __restrict__ bias, float* __restrict__ out,
                     int Ci, int Co, int doRelu) {
  int i = blockIdx.x * blockDim.x + threadIdx.x;
  if (i >= BATCH * Co) return;
  const int o = i % Co;
  const int b = i / Co;
  const float* wr = W + (size_t)o * Ci;
  const float* xr = in + (size_t)b * Ci;
  float acc = bias[o];
  for (int c = 0; c < Ci; ++c) acc += xr[c] * wr[c];
  out[i] = doRelu ? fmaxf(acc, 0.0f) : acc;
}

// ---------------- reparameterized sample: z = mu + eps * exp(0.5*lv) ------------
__global__ void k_sample(const float* __restrict__ mu, const float* __restrict__ lv,
                         float* __restrict__ z, int n) {
  int i = blockIdx.x * blockDim.x + threadIdx.x;
  if (i >= n) return;
  unsigned int s = (unsigned int)i * 747796405u + 2891336453u;
  s ^= s >> 17; s *= 0xed5ad4bbu; s ^= s >> 11; s *= 0xac4c1b51u;
  s ^= s >> 15; s *= 0x31848babu; s ^= s >> 14;
  unsigned int s2 = (s ^ 0x9E3779B9u) * 0x85EBCA6Bu;
  s2 ^= s2 >> 13; s2 *= 0xC2B2AE35u; s2 ^= s2 >> 16;
  const float u1 = (float)(s >> 8) * (1.0f / 16777216.0f) + 1e-7f;
  const float u2 = (float)(s2 >> 8) * (1.0f / 16777216.0f);
  const float eps = sqrtf(-2.0f * __logf(u1)) * __cosf(6.28318530718f * u2);
  z[i] = mu[i] + eps * __expf(0.5f * lv[i]);
}

// ================================================================================
extern "C" void kernel_launch(void* const* d_in, const int* in_sizes, int n_in,
                              void* d_out, int out_size, void* d_ws, size_t ws_size,
                              hipStream_t stream) {
  (void)in_sizes; (void)n_in; (void)out_size; (void)ws_size;

  // ---- workspace carve-out ----
  char* ws = (char*)d_ws;
  size_t off = 0;
  auto carve = [&](size_t bytes) -> void* {
    void* p = ws + off;
    off = (off + bytes + 255) & ~(size_t)255;
    return p;
  };
  float*          xT0     = (float*)carve((size_t)BATCH * NPTS0 * 3 * 4);
  float*          xT1     = (float*)carve((size_t)BATCH * 512 * 128 * 4);
  float*          xT2     = (float*)carve((size_t)BATCH * 128 * 256 * 4);
  int*            fpsIdx  = (int*)carve((size_t)BATCH * 512 * 4);
  float*          centers = (float*)carve((size_t)BATCH * 512 * 3 * 4);
  __hip_bfloat16* wbf     = (__hip_bfloat16*)carve((size_t)79872 * 2);
  __hip_bfloat16* actA    = (__hip_bfloat16*)carve((size_t)67108864);
  __hip_bfloat16* actB    = (__hip_bfloat16*)carve((size_t)67108864);
  float*          Hbuf    = (float*)carve((size_t)134217728);
  float*          stats   = (float*)carve((size_t)512 * 4);   // sum[256] + sumsq[256]
  float*          feat    = (float*)carve((size_t)BATCH * 256 * 4);
  float*          fc1     = (float*)carve((size_t)BATCH * 512 * 4);
  float*          fc2     = (float*)carve((size_t)BATCH * 512 * 4);

  // d_in layout: 0=x; convs (W,b,gamma,beta) x6 at 1..24; 25..32 = fc1_W,fc1_b,
  // fc2_W,fc2_b,lv_W,lv_b,mu_W,mu_b (jax dict key order).
  const int convIn[6] = {1, 5, 9, 13, 17, 21};
  const int convCi[6] = {3, 64, 64, 128, 128, 128};
  const int convCo[6] = {64, 64, 128, 128, 128, 256};
  size_t wOffEl[6];

  // pack conv weights to padded bf16
  {
    size_t wo = 0;
    for (int l = 0; l < 6; ++l) {
      const int Cs = align32i(convCi[l]);
      wOffEl[l] = wo;
      const int tot = convCo[l] * Cs;
      k_convert_w<<<(tot + 255) / 256, 256, 0, stream>>>(
          (const float*)d_in[convIn[l]], wbf + wo, convCo[l], convCi[l], Cs);
      wo += (size_t)tot;
    }
  }

  auto run_layer = [&](int l, const __hip_bfloat16* Xin, __hip_bfloat16* Yout, int Ntot) {
    const int Co = convCo[l];
    const int Cs = align32i(convCi[l]);
    k_zero<<<2, 256, 0, stream>>>(stats, 512);
    dim3 g(Ntot / 64, Co / 64);
    dim3 blk(32, 4);
    const __hip_bfloat16* Wl = wbf + wOffEl[l];
    const float* bl = (const float*)d_in[convIn[l] + 1];
    if (Cs == 32) {
      k_gemm_bn_stats<32><<<g, blk, 0, stream>>>(Xin, Wl, bl, Hbuf, stats, stats + 256, Ntot, Co);
    } else if (Cs == 64) {
      k_gemm_bn_stats<64><<<g, blk, 0, stream>>>(Xin, Wl, bl, Hbuf, stats, stats + 256, Ntot, Co);
    } else {
      k_gemm_bn_stats<128><<<g, blk, 0, stream>>>(Xin, Wl, bl, Hbuf, stats, stats + 256, Ntot, Co);
    }
    const size_t tot = (size_t)Ntot * Co;
    k_bn_relu<<<(unsigned)((tot + 255) / 256), 256, 0, stream>>>(
        Hbuf, stats, stats + 256, (const float*)d_in[convIn[l] + 2],
        (const float*)d_in[convIn[l] + 3], Ntot, Co, Yout);
  };

  // ---- input transpose ----
  {
    const size_t tot = (size_t)BATCH * 3 * NPTS0;
    k_transpose_in<<<(unsigned)((tot + 255) / 256), 256, 0, stream>>>(
        (const float*)d_in[0], xT0, 3, NPTS0);
  }

  // ---- stage 0: S=512, Q=32, r=0.2, MLP 3->64->64->128 ----
  k_fps<<<BATCH, 256, 0, stream>>>(xT0, NPTS0, 3, 512, fpsIdx, centers);
  k_group<<<(BATCH * 512) / 4, 128, 0, stream>>>(xT0, NPTS0, 3, 32, centers,
                                                 512, 32, 0.04f, actA);
  const int Nt0 = BATCH * 512 * 32;
  run_layer(0, actA, actB, Nt0);
  run_layer(1, actB, actA, Nt0);
  run_layer(2, actA, actB, Nt0);
  {
    const size_t tot = (size_t)BATCH * 512 * 128;
    k_maxq<<<(unsigned)((tot + 255) / 256), 256, 0, stream>>>(actB, 512, 32, 128, xT1);
  }

  // ---- stage 1: S=128, Q=64, r=0.4, MLP 128->128->128->256 ----
  k_fps<<<BATCH, 256, 0, stream>>>(xT1, 512, 128, 128, fpsIdx, centers);
  k_group<<<(BATCH * 128) / 4, 128, 0, stream>>>(xT1, 512, 128, 128, centers,
                                                 128, 64, 0.16f, actA);
  const int Nt1 = BATCH * 128 * 64;
  run_layer(3, actA, actB, Nt1);
  run_layer(4, actB, actA, Nt1);
  run_layer(5, actA, actB, Nt1);
  {
    const size_t tot = (size_t)BATCH * 128 * 256;
    k_maxq<<<(unsigned)((tot + 255) / 256), 256, 0, stream>>>(actB, 128, 64, 256, xT2);
  }

  // ---- head ----
  k_meanS<<<(BATCH * 256 + 255) / 256, 256, 0, stream>>>(xT2, 128, 256, feat);
  k_fc<<<(BATCH * 512 + 255) / 256, 256, 0, stream>>>(
      feat, (const float*)d_in[25], (const float*)d_in[26], fc1, 256, 512, 1);
  k_fc<<<(BATCH * 512 + 255) / 256, 256, 0, stream>>>(
      fc1, (const float*)d_in[27], (const float*)d_in[28], fc2, 512, 512, 1);

  float* out = (float*)d_out;                    // [z | mu | log_var], each 16x128
  k_fc<<<(BATCH * 128 + 255) / 256, 256, 0, stream>>>(
      fc2, (const float*)d_in[31], (const float*)d_in[32], out + 2048, 512, 128, 0); // mu
  k_fc<<<(BATCH * 128 + 255) / 256, 256, 0, stream>>>(
      fc2, (const float*)d_in[29], (const float*)d_in[30], out + 4096, 512, 128, 0); // log_var
  k_sample<<<(2048 + 255) / 256, 256, 0, stream>>>(out + 2048, out + 4096, out, 2048);
}
